// SimpleLSTM_42288247996637
// MI455X (gfx1250) — matrix operand order repaired and checked by reading, not compiled, not run on hardware
//
#include <hip/hip_runtime.h>
#include <hip/hip_bf16.h>

// ---------------------------------------------------------------------------
// LSTM (B=64, T=2048, H=512) for gfx1250 / MI455X.
//  - 8 persistent workgroups x 256 threads (8 wave32 waves), one per WGP.
//  - W_hh slice (256 rows x 512 cols) bf16 in WMMA-B tile layout, resident in
//    256KB LDS for the whole kernel.
//  - Recurrent GEMM via v_wmma_f32_16x16x32_bf16, 128 WMMA/wave/step, with
//    batched B-tile LDS loads (8 tiles in flight) and double-buffered A tiles
//    so DS/VMEM latency overlaps the WMMA chain.
//  - Gate nonlinearity + c/h update entirely in registers.
//  - h handed off between steps via a 64KB global bf16 buffer in WMMA-A
//    operand layout; device-scope atomic barrier per step.
// ---------------------------------------------------------------------------

#define T_STEPS 2048
#define BATCH   64
#define HID     512
#define NWG     8
#define TPB     256

typedef __attribute__((ext_vector_type(16))) __bf16        v16bf;
typedef __attribute__((ext_vector_type(8)))  float         v8f;
typedef __attribute__((ext_vector_type(8)))  unsigned int  v8u;

// workspace byte offsets
#define WB_OFF    0u          // 8 * 256KB bf16 W_hh tiles         (2,097,152 B)
#define HBF_OFF   2097152u    // h bf16, WMMA-A layout             (   65,536 B)
#define XT_OFF    2162688u    // x transposed [T][B] fp32          (  524,288 B)
#define HFIN_OFF  2686976u    // final h fp32 [B,H]                (  131,072 B)
#define CFIN_OFF  2818048u    // final c fp32 [B,H]                (  131,072 B)
#define YPART_OFF 2949120u    // y partials [NWG][T][B] fp32       (4,194,304 B)
#define BAR_OFF   7143424u    // barrier counter

__device__ __forceinline__ unsigned short f2bf(float f) {
    unsigned u = __float_as_uint(f);
    u += 0x7FFFu + ((u >> 16) & 1u);   // round-to-nearest-even
    return (unsigned short)(u >> 16);
}

__device__ __forceinline__ v16bf ld32_as_bf16(const void* p) {
    v8u r = *(const v8u*)p;
    return __builtin_bit_cast(v16bf, r);
}

__device__ __forceinline__ float sigmoidf(float z) {
    return 1.0f / (1.0f + __expf(-z));
}

// ---------------------------------------------------------------------------
// prep: W_hh fp32 -> bf16 in per-workgroup WMMA-B tile layout, transpose x,
//       zero the bf16 h buffer (h_{-1}=0) and the barrier counter.
// B-operand tile (32x16 bf16, 1KB): lane L holds column n'=L%16, K values
// (L/16)*16 + p  for p=0..15, packed 2 per dword -> 32 contiguous bytes/lane.
// ---------------------------------------------------------------------------
__global__ void lstm_prep_kernel(const float* __restrict__ Whh,
                                 const float* __restrict__ x,
                                 unsigned short* __restrict__ wb,
                                 unsigned short* __restrict__ hbf,
                                 float* __restrict__ xT,
                                 unsigned* __restrict__ bar) {
    int e = blockIdx.x * TPB + threadIdx.x;          // 0 .. 1,048,575
    int s      = e >> 17;                            // workgroup slice
    int rem    = e & 131071;
    int tile   = rem >> 9;                           // 0..255  (ntile*16 + kt)
    int within = rem & 511;
    int lane   = within >> 4;
    int p      = within & 15;
    int ntile  = tile >> 4;                          // g*4 + jt
    int kt     = tile & 15;
    int g      = ntile >> 2;
    int jt     = ntile & 3;
    int nn     = lane & 15;
    int khalf  = lane >> 4;
    int kk     = khalf * 16 + p;
    int R      = g * HID + s * 64 + jt * 16 + nn;    // W_hh row (gate row)
    int C      = kt * 32 + kk;                       // W_hh col (hidden k)
    wb[e] = f2bf(Whh[R * HID + C]);
    if (e < BATCH * T_STEPS) {                       // x[b][t] -> xT[t][b]
        int b = e >> 11, t = e & (T_STEPS - 1);
        xT[t * BATCH + b] = x[e];
    }
    if (e < 32768) hbf[e] = 0;                       // zero h(-1)
    if (e == 0)    *bar = 0u;
}

// ---------------------------------------------------------------------------
// persistent LSTM kernel: 8 workgroups x 256 threads, loops over 2048 steps.
// ---------------------------------------------------------------------------
__global__ __launch_bounds__(TPB, 1)
void lstm_persistent_kernel(const float* __restrict__ xT,
                            const float* __restrict__ Wih,
                            const float* __restrict__ bih,
                            const float* __restrict__ bhh,
                            const float* __restrict__ Wfc,
                            const unsigned short* __restrict__ wb,
                            unsigned short* __restrict__ hbf,
                            float* __restrict__ hfin,
                            float* __restrict__ cfin,
                            float* __restrict__ ypart,
                            unsigned* __restrict__ bar) {
    extern __shared__ __align__(32) unsigned char smem[];     // 256KB W + 8KB red
    float* red = (float*)(smem + 262144);

    const int s    = blockIdx.x;
    const int tid  = threadIdx.x;
    const int wave = tid >> 5;
    const int lane = tid & 31;
    const int mstrip = wave & 3;      // batch strip (16 rows)
    const int jh     = wave >> 2;     // hidden half (32 units) of this slice
    const int nn     = lane & 15;     // WMMA column within tile
    const int half   = lane >> 4;

    // ---- stage this workgroup's W_hh bf16 tiles into LDS (256KB) ----
    {
        const uint4* src = (const uint4*)(wb + (size_t)s * 131072u);
        uint4*       dst = (uint4*)smem;
        #pragma unroll 4
        for (int i = tid; i < 16384; i += TPB) dst[i] = src[i];
    }
    __syncthreads();

    // ---- per-lane constants (gate order: i, f, g, o) ----
    float wih[4][2], bsum[4][2], wfc[2];
    int   jglob[2], hoff[2];
    #pragma unroll
    for (int jj = 0; jj < 2; ++jj) {
        int jl = (jh * 2 + jj) * 16 + nn;            // 0..63 within slice
        jglob[jj] = s * 64 + jl;                     // global hidden index
        int kth = s * 2 + (jl >> 5);                 // K-tile of jglob
        int kp  = jl & 31;
        int ha  = kp >> 4;
        int pa  = kp & 15;
        hoff[jj] = ((mstrip * 16 + kth) << 10) + ha * 512 + pa * 2;
        wfc[jj] = Wfc[jglob[jj]];
        #pragma unroll
        for (int g = 0; g < 4; ++g) {
            int Rg = g * HID + jglob[jj];
            wih[g][jj]  = Wih[Rg];
            bsum[g][jj] = bih[Rg] + bhh[Rg];
        }
    }

    // LDS byte offsets of this wave's 8 B tiles (ntile stride 16KB, kt stride 1KB)
    int boff[8];
    #pragma unroll
    for (int u = 0; u < 8; ++u) {
        int g = u >> 1, jj = u & 1;
        int ntile = g * 4 + jh * 2 + jj;
        boff[u] = (ntile << 14) + lane * 32;         // + kt*1024 at use site
    }
    const int aoff = (mstrip << 14) + lane * 32;     // + kt*1024 at use site

    float cst[2][8];
    #pragma unroll
    for (int jj = 0; jj < 2; ++jj)
        #pragma unroll
        for (int r = 0; r < 8; ++r) cst[jj][r] = 0.0f;

    unsigned btarget = 0;

    for (int t = 0; t < T_STEPS; ++t) {
        // ---- recurrent GEMM: acc[g*2+jj] = h(t-1) x W_slice^T ----
        v8f acc[8];
        #pragma unroll
        for (int u = 0; u < 8; ++u)
            #pragma unroll
            for (int r = 0; r < 8; ++r) acc[u][r] = 0.0f;

        v16bf a_cur = ld32_as_bf16((const unsigned char*)hbf + aoff);
        #pragma unroll
        for (int kt = 0; kt < 16; ++kt) {
            // batch all 8 B tiles for this k-step (loads stay in flight)
            v16bf bt[8];
            #pragma unroll
            for (int u = 0; u < 8; ++u)
                bt[u] = ld32_as_bf16(smem + boff[u] + (kt << 10));
            // prefetch next A tile under the WMMA chain
            v16bf a_nxt = ld32_as_bf16((const unsigned char*)hbf +
                                       aoff + (((kt + 1) & 15) << 10));
            #pragma unroll
            for (int u = 0; u < 8; ++u)
                acc[u] = __builtin_amdgcn_wmma_f32_16x16x32_bf16(
                    false, a_cur, false, bt[u], (short)0, acc[u], false, false);
            a_cur = a_nxt;
        }

        // ---- input projection + nonlinearity + state update (registers) ----
        float xb[8];
        #pragma unroll
        for (int r = 0; r < 8; ++r)
            xb[r] = xT[t * BATCH + (mstrip * 16 + half * 8 + r)];

        float s8[8];
        #pragma unroll
        for (int r = 0; r < 8; ++r) s8[r] = 0.0f;

        #pragma unroll
        for (int jj = 0; jj < 2; ++jj) {
            #pragma unroll
            for (int r = 0; r < 8; ++r) {
                float ig = sigmoidf(acc[0 * 2 + jj][r] + xb[r] * wih[0][jj] + bsum[0][jj]);
                float fg = sigmoidf(acc[1 * 2 + jj][r] + xb[r] * wih[1][jj] + bsum[1][jj]);
                float gg = tanhf   (acc[2 * 2 + jj][r] + xb[r] * wih[2][jj] + bsum[2][jj]);
                float og = sigmoidf(acc[3 * 2 + jj][r] + xb[r] * wih[3][jj] + bsum[3][jj]);
                float cn = fg * cst[jj][r] + ig * gg;
                cst[jj][r] = cn;
                float hv = og * tanhf(cn);
                // store h as bf16 in WMMA-A layout for next step
                int mp = half * 8 + r;               // m' within tile
                *(unsigned short*)((unsigned char*)hbf + hoff[jj] + mp * 32) = f2bf(hv);
                s8[r] += hv * wfc[jj];
                if (t == T_STEPS - 1) {
                    int b = mstrip * 16 + mp;
                    hfin[b * HID + jglob[jj]] = hv;
                    cfin[b * HID + jglob[jj]] = cn;
                }
            }
        }

        // ---- deterministic y-partial reduction over this slice ----
        #pragma unroll
        for (int r = 0; r < 8; ++r) red[tid * 8 + r] = s8[r];
        __syncthreads();
        if (tid < BATCH) {
            int b  = tid;
            int ms = b >> 4, hf = (b >> 3) & 1, r = b & 7;
            float a = 0.0f;
            #pragma unroll
            for (int j2 = 0; j2 < 2; ++j2) {
                int w = j2 * 4 + ms;
                #pragma unroll
                for (int n2 = 0; n2 < 16; ++n2)
                    a += red[(w * 32 + hf * 16 + n2) * 8 + r];
            }
            ypart[((size_t)s * T_STEPS + t) * BATCH + b] = a;
        }

        // ---- device-scope barrier: publish h(t), wait for all 8 WGPs ----
        btarget += NWG;
        __threadfence();
        __syncthreads();
        if (tid == 0) {
            __hip_atomic_fetch_add(bar, 1u, __ATOMIC_ACQ_REL, __HIP_MEMORY_SCOPE_AGENT);
            while (__hip_atomic_load(bar, __ATOMIC_ACQUIRE, __HIP_MEMORY_SCOPE_AGENT) < btarget)
                __builtin_amdgcn_s_sleep(2);
        }
        __syncthreads();
        __threadfence();
    }
}

// ---------------------------------------------------------------------------
// finalize: y = sum of 8 partials + b_fc ; copy h_n, c_n.
// d_out = [ y (64*2048) | h_n (64*512) | c_n (64*512) ]
// ---------------------------------------------------------------------------
__global__ void lstm_finalize_kernel(const float* __restrict__ ypart,
                                     const float* __restrict__ hfin,
                                     const float* __restrict__ cfin,
                                     const float* __restrict__ bfc,
                                     float* __restrict__ out) {
    int i = blockIdx.x * TPB + threadIdx.x;          // 0 .. 196,607
    if (i < BATCH * T_STEPS) {
        int b = i >> 11, t = i & (T_STEPS - 1);
        float a = bfc[0];
        #pragma unroll
        for (int s2 = 0; s2 < NWG; ++s2)
            a += ypart[((size_t)s2 * T_STEPS + t) * BATCH + b];
        out[i] = a;
    } else if (i < BATCH * T_STEPS + BATCH * HID) {
        out[i] = hfin[i - BATCH * T_STEPS];
    } else if (i < BATCH * T_STEPS + 2 * BATCH * HID) {
        out[i] = cfin[i - (BATCH * T_STEPS + BATCH * HID)];
    }
}

extern "C" void kernel_launch(void* const* d_in, const int* in_sizes, int n_in,
                              void* d_out, int out_size, void* d_ws, size_t ws_size,
                              hipStream_t stream) {
    (void)in_sizes; (void)n_in; (void)out_size; (void)ws_size;
    const float* x   = (const float*)d_in[0];
    const float* Wih = (const float*)d_in[1];
    const float* Whh = (const float*)d_in[2];
    const float* bih = (const float*)d_in[3];
    const float* bhh = (const float*)d_in[4];
    const float* Wfc = (const float*)d_in[5];
    const float* bfc = (const float*)d_in[6];

    unsigned char* ws = (unsigned char*)d_ws;
    unsigned short* wb    = (unsigned short*)(ws + WB_OFF);
    unsigned short* hbf   = (unsigned short*)(ws + HBF_OFF);
    float*          xT    = (float*)(ws + XT_OFF);
    float*          hfin  = (float*)(ws + HFIN_OFF);
    float*          cfin  = (float*)(ws + CFIN_OFF);
    float*          ypart = (float*)(ws + YPART_OFF);
    unsigned*       bar   = (unsigned*)(ws + BAR_OFF);

    const int smemBytes = 262144 + 8192;   // W tiles + reduction scratch
    hipFuncSetAttribute((const void*)lstm_persistent_kernel,
                        hipFuncAttributeMaxDynamicSharedMemorySize, smemBytes);

    lstm_prep_kernel<<<4096, TPB, 0, stream>>>(Whh, x, wb, hbf, xT, bar);
    lstm_persistent_kernel<<<NWG, TPB, smemBytes, stream>>>(
        xT, Wih, bih, bhh, Wfc, wb, hbf, hfin, cfin, ypart, bar);
    lstm_finalize_kernel<<<768, TPB, 0, stream>>>(ypart, hfin, cfin, bfc, (float*)d_out);
}
